// SparseMoE_82514911691454
// MI455X (gfx1250) — matrix operand order repaired and checked by reading, not compiled, not run on hardware
//
#include <hip/hip_runtime.h>
#include <math.h>

#define T_TOK   8192
#define D_MODEL 1024
#define DFF     4096
#define N_EXP   8
#define TOPK    2
#define M_TILE  32          // tokens per workgroup tile
#define FF_CHUNK 512        // d_ff columns per fused chunk

typedef __attribute__((ext_vector_type(16))) __bf16 v16bf;
typedef __attribute__((ext_vector_type(8)))  float  v8f;

// ---------------------------------------------------------------- zero init
__global__ void zero_init_kernel(float* __restrict__ out, int n,
                                 int* __restrict__ cnt, float* __restrict__ gsum) {
  int i = blockIdx.x * blockDim.x + threadIdx.x;
  for (int j = i; j < n; j += gridDim.x * blockDim.x) out[j] = 0.0f;
  if (i < N_EXP) { cnt[i] = 0; gsum[i] = 0.0f; }
}

// ---------------------------------------------------------------- router
__global__ void router_kernel(const float* __restrict__ x, const float* __restrict__ gw,
                              float* __restrict__ w_te, int* __restrict__ cnt,
                              float* __restrict__ gsum, int* __restrict__ lists) {
  int t = blockIdx.x * blockDim.x + threadIdx.x;
  if (t >= T_TOK) return;
  float acc[N_EXP];
#pragma unroll
  for (int e = 0; e < N_EXP; ++e) acc[e] = 0.0f;
  const float* xr = x + (size_t)t * D_MODEL;
  for (int k = 0; k < D_MODEL; ++k) {
    float xv = xr[k];
#pragma unroll
    for (int e = 0; e < N_EXP; ++e) acc[e] += xv * gw[e * D_MODEL + k];
  }
  // top-2 (first index wins ties, matching jax.lax.top_k)
  int i0 = 0;
#pragma unroll
  for (int e = 1; e < N_EXP; ++e) if (acc[e] > acc[i0]) i0 = e;
  int i1 = (i0 == 0) ? 1 : 0;
#pragma unroll
  for (int e = 0; e < N_EXP; ++e) if (e != i0 && acc[e] > acc[i1]) i1 = e;
  float e1 = __expf(acc[i1] - acc[i0]);       // <= 1
  float den = 1.0f + e1;
  float g0 = 1.0f / den;
  float g1 = e1 / den;
#pragma unroll
  for (int e = 0; e < N_EXP; ++e) w_te[t * N_EXP + e] = 0.0f;
  w_te[t * N_EXP + i0] = g0;
  w_te[t * N_EXP + i1] = g1;
  int p0 = atomicAdd(&cnt[i0], 1); lists[i0 * T_TOK + p0] = t;
  int p1 = atomicAdd(&cnt[i1], 1); lists[i1 * T_TOK + p1] = t;
  atomicAdd(&gsum[i0], g0);
  atomicAdd(&gsum[i1], g1);
}

// branch-free GELU (tanh form via sigmoid): 0.5x(1+tanh(.79788(x+.044715x^3)))
__device__ __forceinline__ float gelu_fast(float v) {
  float u = v * (1.5957691f + 0.07135481f * v * v);
  return v / (1.0f + __expf(-u));
}

// ---------------------------------------------------------------- expert FFN
__launch_bounds__(256, 1)
__global__ void moe_kernel(const float* __restrict__ x,
                           const float* __restrict__ w1, const float* __restrict__ b1,
                           const float* __restrict__ w2, const float* __restrict__ b2,
                           const float* __restrict__ w_te, const int* __restrict__ cnt,
                           const int* __restrict__ lists, float* __restrict__ out) {
  const int e = blockIdx.y;
  const int tile = blockIdx.x;
  const int n_tok = cnt[e];
  if (tile * M_TILE >= n_tok) return;   // block-uniform: EXEC all-1 for WMMA

  __shared__ __bf16 xs[M_TILE * D_MODEL];   // 64 KB: x tile, bf16
  __shared__ __bf16 hs[M_TILE * FF_CHUNK];  // 32 KB: gelu(h) chunk, bf16
  __shared__ int    tok_s[M_TILE];
  __shared__ float  gate_s[M_TILE];

  const int tid  = threadIdx.x;
  const int lane = tid & 31;
  const int wave = tid >> 5;
  const int lrow = lane & 15;
  const int lhi  = lane >> 4;

  if (tid < M_TILE) {
    int idx = tile * M_TILE + tid;
    int tok = (idx < n_tok) ? lists[e * T_TOK + idx] : 0;
    tok_s[tid]  = tok;
    gate_s[tid] = (idx < n_tok) ? w_te[tok * N_EXP + e] : 0.0f; // padded rows add 0
  }
  __syncthreads();

  { // stage x tile -> LDS bf16, 128 contiguous floats per thread
    int row = tid >> 3;
    int cb  = (tid & 7) * 128;
    const float* xr = x + (size_t)tok_s[row] * D_MODEL + cb;
#pragma unroll 8
    for (int j = 0; j < 128; ++j) xs[row * D_MODEL + cb + j] = (__bf16)xr[j];
  }
  __syncthreads();

  const float* w1e = w1 + (size_t)e * DFF * D_MODEL;
  const float* b1e = b1 + (size_t)e * DFF;
  const float* w2e = w2 + (size_t)e * D_MODEL * DFF;
  const float* b2e = b2 + (size_t)e * D_MODEL;

  // persistent GEMM2 accumulators: wave owns 128 out cols x 32 rows = 2x8 tiles
  const int col2 = wave * 128;
  v8f acc2[2][8];
#pragma unroll
  for (int mt = 0; mt < 2; ++mt)
#pragma unroll
    for (int tj = 0; tj < 8; ++tj) {
      float b2v = b2e[col2 + tj * 16 + lrow];
#pragma unroll
      for (int r = 0; r < 8; ++r) acc2[mt][tj][r] = b2v;
    }

  for (int chunk = 0; chunk < DFF; chunk += FF_CHUNK) {
    // ---- GEMM1: wave computes h cols [chunk + wave*64, +64) for 32 rows
    const int c0 = wave * 64;          // chunk-local col base
    v8f h1[2][4];
#pragma unroll
    for (int mt = 0; mt < 2; ++mt)
#pragma unroll
      for (int tj = 0; tj < 4; ++tj) {
        float b1v = b1e[chunk + c0 + tj * 16 + lrow];
#pragma unroll
        for (int r = 0; r < 8; ++r) h1[mt][tj][r] = b1v;
      }
    for (int k0 = 0; k0 < D_MODEL; k0 += 32) {
      // A frags (two 16-row tiles), contiguous K: lane half h holds k = 16h + s
      v16bf aA[2];
#pragma unroll
      for (int mt = 0; mt < 2; ++mt) {
        const __bf16* ap = &xs[(mt * 16 + lrow) * D_MODEL + k0 + (lhi ? 16 : 0)];
#pragma unroll
        for (int j = 0; j < 8; ++j) { aA[mt][j] = ap[j]; aA[mt][j + 8] = ap[8 + j]; }
      }
#pragma unroll
      for (int tj = 0; tj < 4; ++tj) {
        // B frag: w1 row (chunk+c0+tj*16+lrow), 16 contiguous K (row-major = K-major)
        const float* bp = w1e + (size_t)(chunk + c0 + tj * 16 + lrow) * D_MODEL
                              + k0 + (lhi ? 16 : 0);
        __builtin_prefetch(bp + 64, 0, 1);
        v16bf bb;
#pragma unroll
        for (int j = 0; j < 16; ++j) bb[j] = (__bf16)bp[j];
#pragma unroll
        for (int mt = 0; mt < 2; ++mt)   // one B fragment feeds both row tiles
          h1[mt][tj] = __builtin_amdgcn_wmma_f32_16x16x32_bf16(
              false, aA[mt], false, bb, (short)0, h1[mt][tj], false, false);
      }
    }
    // GELU (branch-free) + store h chunk; C layout: M = r + 8*lhi, N = lrow
#pragma unroll
    for (int mt = 0; mt < 2; ++mt)
#pragma unroll
      for (int tj = 0; tj < 4; ++tj)
#pragma unroll
        for (int r = 0; r < 8; ++r) {
          int M = mt * 16 + lhi * 8 + r;
          hs[M * FF_CHUNK + c0 + tj * 16 + lrow] = (__bf16)gelu_fast(h1[mt][tj][r]);
        }
    __syncthreads();   // hs chunk complete

    // ---- GEMM2 partial over this chunk's K = FF_CHUNK
    for (int kk = 0; kk < FF_CHUNK; kk += 32) {
      v16bf aA[2];
#pragma unroll
      for (int mt = 0; mt < 2; ++mt) {
        const __bf16* ap = &hs[(mt * 16 + lrow) * FF_CHUNK + kk + (lhi ? 16 : 0)];
#pragma unroll
        for (int j = 0; j < 8; ++j) { aA[mt][j] = ap[j]; aA[mt][j + 8] = ap[8 + j]; }
      }
#pragma unroll
      for (int tj = 0; tj < 8; ++tj) {
        const float* bp = w2e + (size_t)(col2 + tj * 16 + lrow) * DFF
                              + chunk + kk + (lhi ? 16 : 0);
        __builtin_prefetch(bp + 64, 0, 1);
        v16bf bb;
#pragma unroll
        for (int j = 0; j < 16; ++j) bb[j] = (__bf16)bp[j];
#pragma unroll
        for (int mt = 0; mt < 2; ++mt)
          acc2[mt][tj] = __builtin_amdgcn_wmma_f32_16x16x32_bf16(
              false, aA[mt], false, bb, (short)0, acc2[mt][tj], false, false);
      }
    }
    __syncthreads();   // before next chunk overwrites hs
  }

  // epilogue: out[tok, col] += gate[tok] * y  (tokens live in 2 expert lists)
#pragma unroll
  for (int mt = 0; mt < 2; ++mt)
#pragma unroll
    for (int tj = 0; tj < 8; ++tj)
#pragma unroll
      for (int r = 0; r < 8; ++r) {
        int M = mt * 16 + lhi * 8 + r;
        atomicAdd(&out[(size_t)tok_s[M] * D_MODEL + col2 + tj * 16 + lrow],
                  gate_s[M] * acc2[mt][tj][r]);
      }
}

// ---------------------------------------------------------------- aux loss
__global__ void aux_kernel(const int* __restrict__ cnt, const float* __restrict__ gsum,
                           float* __restrict__ out_aux) {
  if (threadIdx.x == 0 && blockIdx.x == 0) {
    float s = 0.0f;
    for (int e = 0; e < N_EXP; ++e)
      s += ((float)cnt[e] / (float)(T_TOK * TOPK)) * (gsum[e] / (float)T_TOK);
    out_aux[0] = 0.01f * (float)N_EXP * s;
  }
}

// ---------------------------------------------------------------- launcher
extern "C" void kernel_launch(void* const* d_in, const int* in_sizes, int n_in,
                              void* d_out, int out_size, void* d_ws, size_t ws_size,
                              hipStream_t stream) {
  const float* x  = (const float*)d_in[0];
  const float* gw = (const float*)d_in[1];
  const float* w1 = (const float*)d_in[2];
  const float* b1 = (const float*)d_in[3];
  const float* w2 = (const float*)d_in[4];
  const float* b2 = (const float*)d_in[5];
  float* out = (float*)d_out;

  // workspace layout: w_te [T*E] f32 | cnt [E] i32 | gsum [E] f32 | lists [E*T] i32
  char*  ws    = (char*)d_ws;
  float* w_te  = (float*)ws;
  int*   cnt   = (int*)  (ws + (size_t)T_TOK * N_EXP * 4);
  float* gsum  = (float*)(ws + (size_t)T_TOK * N_EXP * 4 + 64);
  int*   lists = (int*)  (ws + (size_t)T_TOK * N_EXP * 4 + 128);

  const int n_out_main = T_TOK * D_MODEL;
  zero_init_kernel<<<(n_out_main + 255) / 256, 256, 0, stream>>>(out, n_out_main, cnt, gsum);
  router_kernel<<<T_TOK / 256, 256, 0, stream>>>(x, gw, w_te, cnt, gsum, lists);
  moe_kernel<<<dim3(T_TOK / M_TILE, N_EXP), 256, 0, stream>>>(x, w1, b1, w2, b2,
                                                              w_te, cnt, lists, out);
  aux_kernel<<<1, 32, 0, stream>>>(cnt, gsum, out + (size_t)n_out_main);
}
